// MultiHeadAttention_21835613733376
// MI455X (gfx1250) — matrix operand order repaired and checked by reading, not compiled, not run on hardware
//
#include <hip/hip_runtime.h>
#include <math.h>

#define NUM_HEADS 16
#define KEY_SIZE  64
#define HEAD_SIZE 64
#define DMODEL    1024
#define SEQ       2048
#define BATCH     2

typedef __attribute__((ext_vector_type(16))) __bf16        v16bf;
typedef __attribute__((ext_vector_type(8)))  float         v8f;
typedef __attribute__((ext_vector_type(8)))  unsigned int  v8u;

// f32 -> bf16 round-to-nearest-even (pre-pass / projection store only)
__device__ __forceinline__ unsigned short f2bf(float f) {
  union { float f; unsigned u; } c; c.f = f;
  unsigned r = c.u + 0x7FFFu + ((c.u >> 16) & 1u);
  return (unsigned short)(r >> 16);
}
__device__ __forceinline__ unsigned pack2(float lo, float hi) {
  return (unsigned)f2bf(lo) | ((unsigned)f2bf(hi) << 16);
}

__device__ __forceinline__ v8f wmma_bf16(v16bf a, v16bf b, v8f c) {
  return __builtin_amdgcn_wmma_f32_16x16x32_bf16(false, a, false, b,
                                                 (short)0, c, false, false);
}
__device__ __forceinline__ v8u make_v8u(uint4 a, uint4 b) {
  v8u r;
  r[0] = a.x; r[1] = a.y; r[2] = a.z; r[3] = a.w;
  r[4] = b.x; r[5] = b.y; r[6] = b.z; r[7] = b.w;
  return r;
}

// ---------------------------------------------------------------------------
// Pre-pass 1: f32 rows -> bf16 rows (b128 in, b64 out, RNE). One-time cost.
// ---------------------------------------------------------------------------
__global__ __launch_bounds__(256)
void cvt_rows_kernel(const float4* __restrict__ X, uint2* __restrict__ Y, int n4) {
  const int i = blockIdx.x * 256 + threadIdx.x;
  if (i < n4) {
    const float4 x = X[i];
    Y[i] = make_uint2(pack2(x.x, x.y), pack2(x.z, x.w));
  }
}

// ---------------------------------------------------------------------------
// Pre-pass 2: W[K,N] f32 -> pair-packed bf16 dwords WP[K/2,N]:
//   WP[k][n] = { bf16(W[2k][n]) , bf16(W[2k+1][n]) }  == B-operand (k,k+1) pair
// ---------------------------------------------------------------------------
__global__ __launch_bounds__(256)
void pack_w_kernel(const float* __restrict__ W, unsigned int* __restrict__ WP,
                   int K, int N) {
  const int i = blockIdx.x * 256 + threadIdx.x;
  const int n4t = N >> 2;
  const int kp = i / n4t;
  const int n0 = (i % n4t) << 2;
  if (kp >= (K >> 1)) return;
  const float4 r0 = *(const float4*)(W + (size_t)(2 * kp) * N + n0);
  const float4 r1 = *(const float4*)(W + (size_t)(2 * kp + 1) * N + n0);
  uint4 o;
  o.x = pack2(r0.x, r1.x); o.y = pack2(r0.y, r1.y);
  o.z = pack2(r0.z, r1.z); o.w = pack2(r0.w, r1.w);
  *(uint4*)(WP + (size_t)kp * N + n0) = o;
}

// ---------------------------------------------------------------------------
// Projection: Y = scale * (X @ W). X bf16 rows (dwords), W pair-packed dwords.
// One wave = 16x64 output tile, K-step 32, no conversions in the hot loop.
// PACKV: store output pair-packed (for V) instead of row-major bf16.
// ---------------------------------------------------------------------------
template <bool PACKV>
__global__ __launch_bounds__(256)
void proj_kernel(const unsigned int* __restrict__ X32,  // bf16 rows, stride K/2 dwords
                 const unsigned int* __restrict__ WP,   // (K/2) x N dwords
                 unsigned short* __restrict__ Y,        // row-major bf16 (!PACKV)
                 unsigned int* __restrict__ YP,         // pair-packed (PACKV)
                 int M, int K, int N, float scale) {
  const int lane = threadIdx.x & 31;
  const int wid  = threadIdx.x >> 5;
  const int gw   = blockIdx.x * 8 + wid;
  const int colTiles = N >> 6;
  const int rowTile  = gw / colTiles;
  const int colTile  = gw % colTiles;
  if (rowTile * 16 >= M) return;
  const int row0 = rowTile * 16;
  const int col0 = colTile * 64;
  const int half = lane >> 4;
  const int mn   = lane & 15;
  const int Kd   = K >> 1;

  v8f acc[4] = {v8f{}, v8f{}, v8f{}, v8f{}};

  const unsigned int* xrow32 = X32 + (size_t)(row0 + mn) * Kd;

  for (int kk = 0; kk < K; kk += 32) {
    // A fragment: dwords 0..7 + 8*half of this 16-dword k-slab
    const uint4* xr = (const uint4*)(xrow32 + (kk >> 1));
    const v8u au = make_v8u(xr[2 * half], xr[2 * half + 1]);
    const v16bf a = __builtin_bit_cast(v16bf, au);

    const unsigned int* wrow = WP + (size_t)(kk >> 1) * N + col0 + mn;
#pragma unroll
    for (int t = 0; t < 4; ++t) {
      v8u bu = {};
#pragma unroll
      for (int r = 0; r < 8; ++r)
        bu[r] = wrow[(size_t)(r + 8 * half) * N + t * 16];
      acc[t] = wmma_bf16(a, __builtin_bit_cast(v16bf, bu), acc[t]);
    }
  }

  if constexpr (PACKV) {
    // pair-pack rows: lane holds both m and m+1 for even r  (m = r + 8*half)
#pragma unroll
    for (int t = 0; t < 4; ++t)
#pragma unroll
      for (int r = 0; r < 8; r += 2) {
        const int m = r + 8 * half;
        YP[(size_t)((row0 + m) >> 1) * N + col0 + t * 16 + mn] =
            pack2(acc[t][r] * scale, acc[t][r + 1] * scale);
      }
  } else {
#pragma unroll
    for (int t = 0; t < 4; ++t)
#pragma unroll
      for (int r = 0; r < 8; ++r) {
        const int m = r + 8 * half;
        Y[(size_t)(row0 + m) * N + col0 + t * 16 + mn] = f2bf(acc[t][r] * scale);
      }
  }
}

// ---------------------------------------------------------------------------
// Flash attention (base-2 softmax; Q pre-scaled by 0.125*log2(e)).
// One wave = one 16-row Q tile; 4 waves / block. Per 64-key chunk:
// 8 WMMAs (S = Q K^T), one f32 max butterfly, one corr/rescale pass,
// LDS transpose of P, 8 WMMAs (O += P V) + 2 WMMAs (row sums via ones-B).
// ---------------------------------------------------------------------------
__global__ __launch_bounds__(128)
void attn_kernel(const unsigned int* __restrict__ Q32,  // bf16 rows, stride 512 dwords
                 const unsigned int* __restrict__ K32,
                 const unsigned int* __restrict__ VP,   // pair-packed: [(B*SEQ)/2, DMODEL]
                 const int* __restrict__ vmask,
                 float* __restrict__ Out) {
  __shared__ __align__(16) unsigned short ldsP[4][16 * 64];

  const int lane = threadIdx.x & 31;
  const int w    = threadIdx.x >> 5;
  const int qBlocks = SEQ / 64;                       // 32
  const int bid  = blockIdx.x;
  const int qblk = bid % qBlocks;
  const int h    = (bid / qBlocks) % NUM_HEADS;
  const int b    = bid / (qBlocks * NUM_HEADS);
  const int sq0  = qblk * 64 + w * 16;
  const int half = lane >> 4;
  const int mn   = lane & 15;
  const int DW   = DMODEL >> 1;                       // 512 dwords per row
  const int hOff = h * (HEAD_SIZE >> 1);              // 32 dwords

  const unsigned int* Qb = Q32 + (size_t)(b * SEQ + sq0) * DW + hOff;
  const unsigned int* Kb = K32 + (size_t)(b * SEQ) * DW + hOff;
  const unsigned int* Vb = VP + (size_t)((b * SEQ) >> 1) * DMODEL + h * HEAD_SIZE;
  const int* mbase = vmask + b * SEQ;

  // Q tile in A layout (resident): d 0..31 and d 32..63
  const uint4* qr = (const uint4*)(Qb + (size_t)mn * DW);
  const v16bf qa0 = __builtin_bit_cast(v16bf, make_v8u(qr[2*half],     qr[2*half + 1]));
  const v16bf qa1 = __builtin_bit_cast(v16bf, make_v8u(qr[4 + 2*half], qr[5 + 2*half]));

  // all-ones B operand for row sums (bf16 1.0 = 0x3F80)
  v8u onesU;
#pragma unroll
  for (int i = 0; i < 8; ++i) onesU[i] = 0x3F803F80u;
  const v16bf onesB = __builtin_bit_cast(v16bf, onesU);

  v8f o[4] = {v8f{}, v8f{}, v8f{}, v8f{}};
  v8f o4 = {};                                        // running row sums
  float mrow[8];
#pragma unroll
  for (int r = 0; r < 8; ++r) mrow[r] = -1e30f;

  for (int sk = 0; sk < SEQ; sk += 64) {
    // ---- scores for 4 key groups of 16 (K as B operand, b128 fragments) ----
    v8f s[4];
#pragma unroll
    for (int g = 0; g < 4; ++g) {
      const uint4* kr = (const uint4*)(Kb + (size_t)(sk + 16 * g + mn) * DW);
      const v16bf kg0 = __builtin_bit_cast(v16bf, make_v8u(kr[2*half],     kr[2*half + 1]));
      const v16bf kg1 = __builtin_bit_cast(v16bf, make_v8u(kr[4 + 2*half], kr[5 + 2*half]));
      v8f sg = {};
      sg   = wmma_bf16(qa0, kg0, sg);
      s[g] = wmma_bf16(qa1, kg1, sg);
    }

    // ---- additive mask (per key column) ----
    float add[4];
#pragma unroll
    for (int g = 0; g < 4; ++g)
      add[g] = mbase[sk + 16 * g + mn] ? 0.0f : -1.0e12f;

    // ---- per-row chunk max: 64 keys -> one butterfly over the 16-lane half ----
    float cm[8];
#pragma unroll
    for (int r = 0; r < 8; ++r) {
      const float c0 = fmaxf(s[0][r] + add[0], s[1][r] + add[1]);
      const float c1 = fmaxf(s[2][r] + add[2], s[3][r] + add[3]);
      float cmv = fmaxf(c0, c1);
#pragma unroll
      for (int off = 1; off < 16; off <<= 1)
        cmv = fmaxf(cmv, __shfl_xor(cmv, off, 32));
      cm[r] = cmv;
    }

    // ---- online update + write P (bf16, truncated) to LDS ----
    unsigned short* lp = ldsP[w];
#pragma unroll
    for (int r = 0; r < 8; ++r) {
      const float mNew = fmaxf(mrow[r], cm[r]);
      const float corr = exp2f(mrow[r] - mNew);
      mrow[r] = mNew;
      const int m = r + 8 * half;
#pragma unroll
      for (int g = 0; g < 4; ++g) {
        const float e = exp2f(s[g][r] + (add[g] - mNew));
        lp[m * 64 + 16 * g + mn] =
            (unsigned short)(__builtin_bit_cast(unsigned, e) >> 16);
      }
      o4[r] *= corr;
#pragma unroll
      for (int t = 0; t < 4; ++t) o[t][r] *= corr;
    }
    __syncthreads();
    const uint4* lp4 = (const uint4*)lp;
    const v16bf pa0 = __builtin_bit_cast(
        v16bf, make_v8u(lp4[mn * 8 + 2 * half],     lp4[mn * 8 + 1 + 2 * half]));
    const v16bf pa1 = __builtin_bit_cast(
        v16bf, make_v8u(lp4[mn * 8 + 4 + 2 * half], lp4[mn * 8 + 5 + 2 * half]));
    __syncthreads();

    // ---- O += P V (pair-packed V), row sums += P @ ones ----
    const unsigned int* vrow = Vb + (size_t)(sk >> 1) * DMODEL + mn;
#pragma unroll
    for (int t = 0; t < 4; ++t) {
      v8u vu0 = {}, vu1 = {};
#pragma unroll
      for (int r = 0; r < 8; ++r) {
        vu0[r] = vrow[(size_t)(r + 8 * half) * DMODEL + t * 16];
        vu1[r] = vrow[(size_t)(16 + r + 8 * half) * DMODEL + t * 16];
      }
      o[t] = wmma_bf16(pa0, __builtin_bit_cast(v16bf, vu0), o[t]);
      o[t] = wmma_bf16(pa1, __builtin_bit_cast(v16bf, vu1), o[t]);
    }
    o4 = wmma_bf16(pa0, onesB, o4);
    o4 = wmma_bf16(pa1, onesB, o4);
  }

  // ---- normalize and store f32 ----
  float* orow = Out + (size_t)(b * SEQ + sq0) * DMODEL + h * HEAD_SIZE;
#pragma unroll
  for (int r = 0; r < 8; ++r) {
    const int m = r + 8 * half;
    const float inv = 1.0f / o4[r];
#pragma unroll
    for (int t = 0; t < 4; ++t)
      orow[(size_t)m * DMODEL + t * 16 + mn] = o[t][r] * inv;
  }
}

// ---------------------------------------------------------------------------
extern "C" void kernel_launch(void* const* d_in, const int* in_sizes, int n_in,
                              void* d_out, int out_size, void* d_ws, size_t ws_size,
                              hipStream_t stream) {
  const float* q     = (const float*)d_in[0];
  const float* k     = (const float*)d_in[1];
  const float* v     = (const float*)d_in[2];
  const int*   vmask = (const int*)d_in[3];
  const float* Wq    = (const float*)d_in[4];
  const float* Wk    = (const float*)d_in[5];
  const float* Wv    = (const float*)d_in[6];

  const int M = BATCH * SEQ;                           // 4096
  const size_t projBytes = (size_t)M * DMODEL * 2;     // 8 MB per tensor (bf16)

  char* ws = (char*)d_ws;
  unsigned short* qw  = (unsigned short*)(ws);                     //  8 MB
  unsigned short* kw  = (unsigned short*)(ws + projBytes);         //  8 MB
  unsigned int*   vp  = (unsigned int*)  (ws + 2 * projBytes);     //  8 MB
  unsigned int*   xbf = (unsigned int*)  (ws + 3 * projBytes);     //  8 MB scratch (reused x3)
  unsigned int*   wp  = (unsigned int*)  (ws + 4 * projBytes);     //  2 MB scratch (reused x3)

  const int n4 = (M * DMODEL) / 4;                     // 1M float4 per tensor
  dim3 cb(256), cg((n4 + 255) / 256);
  const int wpThreads = (DMODEL / 2) * (DMODEL / 4);   // 131072
  dim3 wg((wpThreads + 255) / 256);

  const int totalWaves = (M / 16) * (DMODEL / 64);     // 4096 waves
  dim3 pb(256), pg((totalWaves + 7) / 8);              // 512 blocks

  // 1/sqrt(KEY_SIZE) * log2(e): folded into Q so attention uses exp2 directly
  const float qScale = 0.125f * 1.44269504088896340736f;

  // Q projection
  cvt_rows_kernel<<<cg, cb, 0, stream>>>((const float4*)q, (uint2*)xbf, n4);
  pack_w_kernel<<<wg, cb, 0, stream>>>(Wq, wp, DMODEL, DMODEL);
  proj_kernel<false><<<pg, pb, 0, stream>>>(xbf, wp, qw, nullptr, M, DMODEL, DMODEL, qScale);
  // K projection
  cvt_rows_kernel<<<cg, cb, 0, stream>>>((const float4*)k, (uint2*)xbf, n4);
  pack_w_kernel<<<wg, cb, 0, stream>>>(Wk, wp, DMODEL, DMODEL);
  proj_kernel<false><<<pg, pb, 0, stream>>>(xbf, wp, kw, nullptr, M, DMODEL, DMODEL, 1.0f);
  // V projection (pair-packed output for the PV B-operand)
  cvt_rows_kernel<<<cg, cb, 0, stream>>>((const float4*)v, (uint2*)xbf, n4);
  pack_w_kernel<<<wg, cb, 0, stream>>>(Wv, wp, DMODEL, DMODEL);
  proj_kernel<true><<<pg, pb, 0, stream>>>(xbf, wp, nullptr, vp, M, DMODEL, DMODEL, 1.0f);

  dim3 ab(128);
  dim3 ag(BATCH * NUM_HEADS * (SEQ / 64));             // 1024 blocks
  attn_kernel<<<ag, ab, 0, stream>>>((const unsigned int*)qw,
                                     (const unsigned int*)kw,
                                     vp, vmask, (float*)d_out);
}
